// EmbedMatcher_54417235640962
// MI455X (gfx1250) — compile-verified, matrix operand order checked by reading, not run on hardware
//
#include <hip/hip_runtime.h>
#include <hip/hip_bf16.h>
#include <math.h>

// EmbedMatcher for MI455X (gfx1250, wave32, WMMA).
// Pipeline: gather/top-k/mean (HBM bound) -> f16 WMMA GEMMs (LDS double-
// buffered, clause-batched unpredicated staging) -> LayerNorm/LSTM -> cosine.
// Algebraic optimizations: mean-before-linear GCN, hoisted w_ih GEMM,
// attn==1 (softmax over singleton) => r == support_g, step-0 GEMM elided.

typedef __attribute__((ext_vector_type(16))) _Float16 v16h;
typedef __attribute__((ext_vector_type(8)))  float    v8f;

#define DEV static __device__ __forceinline__

DEV float warp_sum(float v) {
#pragma unroll
  for (int m = 16; m; m >>= 1) v += __shfl_xor(v, m, 32);
  return v;
}

DEV float block_sum(float v, float* red) {
  v = warp_sum(v);
  int lane = threadIdx.x & 31, wid = threadIdx.x >> 5;
  if (lane == 0) red[wid] = v;
  __syncthreads();
  float r = 0.f;
  int nw = (blockDim.x + 31) >> 5;
  if (threadIdx.x < nw) r = red[threadIdx.x];
  if (wid == 0) r = warp_sum(r);
  if (threadIdx.x == 0) red[0] = r;
  __syncthreads();
  r = red[0];
  __syncthreads();
  return r;
}

DEV float sigmoidf(float x) { return 1.f / (1.f + expf(-x)); }

// ---------------------------------------------------------------- utilities
__global__ void cvt_f32_to_f16_kernel(const float* __restrict__ in,
                                      _Float16* __restrict__ out, int n) {
  int i = blockIdx.x * blockDim.x + threadIdx.x;
  if (i < n) out[i] = (_Float16)in[i];
}

__global__ void vec_add_kernel(const float* __restrict__ a,
                               const float* __restrict__ b,
                               float* __restrict__ o, int n) {
  int i = blockIdx.x * blockDim.x + threadIdx.x;
  if (i < n) o[i] = a[i] + b[i];
}

__global__ void zero_kernel(float* __restrict__ p, int n) {
  int i = blockIdx.x * blockDim.x + threadIdx.x;
  if (i < n) p[i] = 0.f;
}

// -------------------------------------------------- neighbor encoder (gather)
// One block per node. 256 threads. Computes mean over exact top-K=32 (by
// cosine sim, stable tie-break like lax.top_k) of [rel | ent] embeddings.
__global__ __launch_bounds__(256) void neighbor_mean_kernel(
    const int* __restrict__ conn,          // (R, 64, 2)
    const int* __restrict__ idsBase, int idStride, int idOff,
    const float* __restrict__ emb,         // (V+1, 256)
    float* __restrict__ outCat)            // (R, 512)
{
  __shared__ float center[256];
  __shared__ float sim[64];
  __shared__ int   sel[64];
  __shared__ float red[8];
  int r = blockIdx.x;
  int tid = threadIdx.x;
  int id = idsBase[r * idStride + idOff];
  float cv = emb[(size_t)id * 256 + tid];
  center[tid] = cv;
  float cn = sqrtf(block_sum(cv * cv, red));   // block_sum syncs -> center visible

  int lane = tid & 31, wid = tid >> 5;
  // 8 waves x 8 iterations cover the 64 neighbors; float4 dot over D=256.
  for (int i = 0; i < 8; i++) {
    int n = i * 8 + wid;
    int entId = conn[((size_t)r * 64 + n) * 2 + 1];
    const float4* e4 = (const float4*)(emb + (size_t)entId * 256);
    const float4* c4 = (const float4*)center;
    float dot = 0.f, n2 = 0.f;
#pragma unroll
    for (int t = 0; t < 2; t++) {
      float4 ev = e4[lane + t * 32];          // lane*16B: conflict-free b128
      float4 cw = c4[lane + t * 32];
      dot += cw.x * ev.x + cw.y * ev.y + cw.z * ev.z + cw.w * ev.w;
      n2  += ev.x * ev.x + ev.y * ev.y + ev.z * ev.z + ev.w * ev.w;
    }
    dot = warp_sum(dot);
    n2  = warp_sum(n2);
    if (lane == 0) sim[n] = dot / fmaxf(cn * sqrtf(n2), 1e-8f);
  }
  __syncthreads();
  // exact top-32 selection via rank count (N=64: 4096 cmps, trivial)
  if (tid < 64) {
    float sv = sim[tid];
    int cnt = 0;
    for (int j = 0; j < 64; j++) {
      float sj = sim[j];
      cnt += (sj > sv) || (sj == sv && j < tid);
    }
    sel[tid] = (cnt < 32);
  }
  __syncthreads();
  // mean over selected; linearity lets the GCN GEMM run on the mean directly.
  float racc = 0.f, eacc = 0.f;
  for (int n = 0; n < 64; n++) {
    if (sel[n]) {
      int relId = conn[((size_t)r * 64 + n) * 2 + 0];
      int entId = conn[((size_t)r * 64 + n) * 2 + 1];
      racc += emb[(size_t)relId * 256 + tid];   // 1KB coalesced per row
      eacc += emb[(size_t)entId * 256 + tid];
    }
  }
  outCat[(size_t)r * 512 + tid]       = racc * (1.f / 32.f);
  outCat[(size_t)r * 512 + 256 + tid] = eacc * (1.f / 32.f);
}

// ------------------------------------------------------------- WMMA GEMM NT
// C[M,N] = act( A[MxK](f32) * W[NxK]^T(f16) + bias[N] + resid[M,N] )
// Block tile 128x128, 8 waves (4x2), wave tile 32x64 = 2x4 x (16x16x32 WMMA)
// => 8 WMMAs per 6 fragment loads per K-tile (vs 4:4 at 32x32).
// Constraints exploited: K % 32 == 0, N % 128 == 0 (true for every GEMM
// here); only M may be ragged. Interior blocks take a branch-free
// unpredicated staging path (global b128 loads issue in one clause); ragged
// blocks use address-clamped loads + post-load select (no exec masking).
#define BM 128
#define BN 128
#define BK 32
#define LDS_S 40   // halfs; 80B rows => every row 16B aligned, frag reads 4B

// Build a 16x32 f16 A/B fragment from a row-major LDS tile, following the
// documented CDNA5 16-bit A layout: lanes 0-15 row m, K{0..7,16..23};
// lanes 16-31 row m, K{8..15,24..31}; 2 halfs per VGPR.
DEV v16h load_frag16x32(const _Float16* tileBase) {
  int lane = threadIdx.x & 31;
  int r  = lane & 15;
  int hs = lane >> 4;
  const unsigned* rowp = (const unsigned*)(tileBase + r * LDS_S);
  union { v16h h; unsigned u[8]; } f;
  int b0 = hs * 4;       // uint index of k = hs*8
  int b1 = 8 + hs * 4;   // uint index of k = 16 + hs*8
  f.u[0] = rowp[b0 + 0]; f.u[1] = rowp[b0 + 1];
  f.u[2] = rowp[b0 + 2]; f.u[3] = rowp[b0 + 3];
  f.u[4] = rowp[b1 + 0]; f.u[5] = rowp[b1 + 1];
  f.u[6] = rowp[b1 + 2]; f.u[7] = rowp[b1 + 3];
  return f.h;
}

// Stage a 128x32 f32 tile as f16 into LDS (2x global b128 -> 1x ds b128).
DEV void stage_a(const float* __restrict__ A, int lda, int M, int blockM,
                 int k0, _Float16* __restrict__ As) {
  int tid = threadIdx.x;
  float4 v[2][2];
  if (blockM + BM <= M) {
    // interior: unconditional loads, all issue before any wait
#pragma unroll
    for (int it = 0; it < 2; it++) {
      int i = tid + it * 256;
      int r = i >> 2, seg = i & 3;                 // 4x 8-float segs per row
      const float* src = A + (size_t)(blockM + r) * lda + k0 + seg * 8;
      v[it][0] = *(const float4*)(src);
      v[it][1] = *(const float4*)(src + 4);
    }
  } else {
    // ragged M: clamp address (always valid), zero-select after load
#pragma unroll
    for (int it = 0; it < 2; it++) {
      int i = tid + it * 256;
      int r = i >> 2, seg = i & 3;
      int m = blockM + r;
      int mc = (m < M) ? m : (M - 1);
      const float* src = A + (size_t)mc * lda + k0 + seg * 8;
      float4 a = *(const float4*)(src);
      float4 b = *(const float4*)(src + 4);
      if (m >= M) {
        a = make_float4(0.f, 0.f, 0.f, 0.f);
        b = make_float4(0.f, 0.f, 0.f, 0.f);
      }
      v[it][0] = a; v[it][1] = b;
    }
  }
#pragma unroll
  for (int it = 0; it < 2; it++) {
    int i = tid + it * 256;
    int r = i >> 2, seg = i & 3;
    union { _Float16 h[8]; uint4 u; } p;
    p.h[0] = (_Float16)v[it][0].x; p.h[1] = (_Float16)v[it][0].y;
    p.h[2] = (_Float16)v[it][0].z; p.h[3] = (_Float16)v[it][0].w;
    p.h[4] = (_Float16)v[it][1].x; p.h[5] = (_Float16)v[it][1].y;
    p.h[6] = (_Float16)v[it][1].z; p.h[7] = (_Float16)v[it][1].w;
    *(uint4*)(As + r * LDS_S + seg * 8) = p.u;     // ds_store_b128, 16B align
  }
}

// Stage a 128x32 f16 weight tile (16B vector copies) into LDS.
DEV void stage_b(const _Float16* __restrict__ W, int ldb, int N, int blockN,
                 int k0, _Float16* __restrict__ Bs) {
  int tid = threadIdx.x;
  uint4 v[2];
  if (blockN + BN <= N) {
#pragma unroll
    for (int it = 0; it < 2; it++) {
      int i = tid + it * 256;
      int r = i >> 2, cv = i & 3;                  // 4x 8-half segs per row
      v[it] = *(const uint4*)(W + (size_t)(blockN + r) * ldb + k0 + cv * 8);
    }
  } else {
#pragma unroll
    for (int it = 0; it < 2; it++) {
      int i = tid + it * 256;
      int r = i >> 2, cv = i & 3;
      int n = blockN + r;
      int nc = (n < N) ? n : (N - 1);
      uint4 t = *(const uint4*)(W + (size_t)nc * ldb + k0 + cv * 8);
      if (n >= N) t = make_uint4(0u, 0u, 0u, 0u);
      v[it] = t;
    }
  }
#pragma unroll
  for (int it = 0; it < 2; it++) {
    int i = tid + it * 256;
    int r = i >> 2, cv = i & 3;
    *(uint4*)(Bs + r * LDS_S + cv * 8) = v[it];    // ds_store_b128
  }
}

__global__ __launch_bounds__(256) void gemm_nt_wmma_kernel(
    const float* __restrict__ A, int lda,
    const _Float16* __restrict__ W, int ldb,
    float* __restrict__ C, int ldc,
    const float* __restrict__ bias,
    const float* __restrict__ resid, int ldr,
    int M, int N, int Kd, int act)   // act: 0 none, 1 relu, 2 tanh
{
  __shared__ _Float16 As[2][BM * LDS_S];
  __shared__ _Float16 Bs[2][BN * LDS_S];
  int tid = threadIdx.x;
  int blockM = blockIdx.y * BM;
  int blockN = blockIdx.x * BN;
  int wid = tid >> 5;
  int wm = wid >> 1, wn = wid & 1;   // 4 waves in M x 2 waves in N

  v8f zero = {};
  v8f acc[2][4];
#pragma unroll
  for (int i = 0; i < 2; i++)
#pragma unroll
    for (int j = 0; j < 4; j++) acc[i][j] = zero;

  // prologue: stage first K tile into buffer 0
  stage_a(A, lda, M, blockM, 0, As[0]);
  stage_b(W, ldb, N, blockN, 0, Bs[0]);

  int buf = 0;
  for (int k0 = 0; k0 < Kd; k0 += BK, buf ^= 1) {
    __syncthreads();   // staged data for `buf` visible; prev reads done

    // prefetch tile-after-next for both streams (global_prefetch_b8)
    if (k0 + 2 * BK < Kd) {
      if (tid < BM) {
        int m = blockM + tid;
        if (m < M) __builtin_prefetch(&A[(size_t)m * lda + k0 + 2 * BK], 0, 3);
      } else {
        int n = blockN + (tid - BM);
        if (n < N) __builtin_prefetch(&W[(size_t)n * ldb + k0 + 2 * BK], 0, 3);
      }
    }

    v16h af[2], bf[4];
    af[0] = load_frag16x32(&As[buf][(wm * 32 +  0) * LDS_S]);
    af[1] = load_frag16x32(&As[buf][(wm * 32 + 16) * LDS_S]);
#pragma unroll
    for (int tn = 0; tn < 4; tn++)
      bf[tn] = load_frag16x32(&Bs[buf][(wn * 64 + tn * 16) * LDS_S]);
#pragma unroll
    for (int tm = 0; tm < 2; tm++)
#pragma unroll
      for (int tn = 0; tn < 4; tn++)
        acc[tm][tn] = __builtin_amdgcn_wmma_f32_16x16x32_f16(
            false, af[tm], false, bf[tn], (short)0, acc[tm][tn], false, false);

    // stage next K tile into the other buffer (overlaps with WMMA above;
    // the barrier at the top of the next iteration orders visibility)
    if (k0 + BK < Kd) {
      stage_a(A, lda, M, blockM, k0 + BK, As[buf ^ 1]);
      stage_b(W, ldb, N, blockN, k0 + BK, Bs[buf ^ 1]);
    }
  }

  // epilogue: C/D layout => VGPR v, lanes<16: M=v, lanes>=16: M=v+8; N=lane&15
  int lane = tid & 31;
  int nl = lane & 15;
  int mh = lane >> 4;
#pragma unroll
  for (int tm = 0; tm < 2; tm++) {
#pragma unroll
    for (int tn = 0; tn < 4; tn++) {
      int nCol = blockN + wn * 64 + tn * 16 + nl;
#pragma unroll
      for (int v = 0; v < 8; v++) {
        int mRow = blockM + wm * 32 + tm * 16 + mh * 8 + v;
        if (mRow < M && nCol < N) {
          float x = acc[tm][tn][v];
          if (bias)  x += bias[nCol];
          if (resid) x += resid[(size_t)mRow * ldr + nCol];
          if (act == 1) x = fmaxf(x, 0.f);
          else if (act == 2) x = tanhf(x);
          C[(size_t)mRow * ldc + nCol] = x;
        }
      }
    }
  }
}

// ------------------------------------------------------------- elementwise
__global__ void layernorm_kernel(const float* __restrict__ X,
                                 float* __restrict__ Y,
                                 const float* __restrict__ g,
                                 const float* __restrict__ b, int Ncols) {
  __shared__ float red[8];
  int row = blockIdx.x;
  const float* x = X + (size_t)row * Ncols;
  float s = 0.f;
  for (int i = threadIdx.x; i < Ncols; i += blockDim.x) s += x[i];
  float mean = block_sum(s, red) / (float)Ncols;
  float s2 = 0.f;
  for (int i = threadIdx.x; i < Ncols; i += blockDim.x) {
    float d = x[i] - mean; s2 += d * d;
  }
  float var = block_sum(s2, red) / (float)Ncols;
  float inv = rsqrtf(var + 1e-5f);
  float* y = Y + (size_t)row * Ncols;
  for (int i = threadIdx.x; i < Ncols; i += blockDim.x)
    y[i] = (x[i] - mean) * inv * g[i] + b[i];
}

__global__ void mean_rows_kernel(const float* __restrict__ X,
                                 float* __restrict__ out, int R, int Ncols) {
  int d = blockIdx.x * blockDim.x + threadIdx.x;
  if (d >= Ncols) return;
  float s = 0.f;
  for (int r = 0; r < R; r++) s += X[(size_t)r * Ncols + d];
  out[d] = s / (float)R;
}

// LSTM cell + h/h_r update. attn==softmax over a single support column == 1,
// so r == support_g exactly; h_r = [h | support_g].
__global__ void lstm_cell_kernel(const float* __restrict__ gates,
                                 const float* __restrict__ query_g,
                                 const float* __restrict__ support_g,
                                 float* __restrict__ c,
                                 float* __restrict__ h,
                                 float* __restrict__ h_r, int Bq) {
  int idx = blockIdx.x * blockDim.x + threadIdx.x;
  if (idx >= Bq * 1024) return;
  int b = idx >> 10, j = idx & 1023;
  const float* gr = gates + (size_t)b * 4096;
  float ig = sigmoidf(gr[j]);
  float fg = sigmoidf(gr[1024 + j]);
  float gg = tanhf(gr[2048 + j]);
  float og = sigmoidf(gr[3072 + j]);
  float cn = fg * c[idx] + ig * gg;
  c[idx] = cn;
  float hh = og * tanhf(cn);
  if (j < 512) {
    float hv = query_g[(size_t)b * 512 + j] + hh;
    h[(size_t)b * 512 + j] = hv;
    h_r[(size_t)b * 1024 + j] = hv;
  } else {
    h_r[(size_t)b * 1024 + j] = support_g[j - 512];
  }
}

__global__ __launch_bounds__(256) void final_dot_kernel(
    const float* __restrict__ h, const float* __restrict__ sg,
    float* __restrict__ out) {
  __shared__ float red[8];
  int b = blockIdx.x;
  float d = 0.f, nh = 0.f, ns = 0.f;
  for (int i = threadIdx.x; i < 512; i += blockDim.x) {
    float hv = h[(size_t)b * 512 + i], sv = sg[i];
    d += hv * sv; nh += hv * hv; ns += sv * sv;
  }
  d  = block_sum(d, red);
  nh = block_sum(nh, red);
  ns = block_sum(ns, red);
  if (threadIdx.x == 0)
    out[b] = d / (fmaxf(sqrtf(nh), 1e-12f) * fmaxf(sqrtf(ns), 1e-12f));
}

// --------------------------------------------------------------- host side
static void launch_gemm(const float* A, int lda, const _Float16* W, int ldb,
                        float* C, int ldc, const float* bias,
                        const float* resid, int ldr,
                        int M, int N, int K, int act, hipStream_t s) {
  dim3 grid((N + BN - 1) / BN, (M + BM - 1) / BM);
  gemm_nt_wmma_kernel<<<grid, 256, 0, s>>>(A, lda, W, ldb, C, ldc, bias,
                                           resid, ldr, M, N, K, act);
}

extern "C" void kernel_launch(void* const* d_in, const int* in_sizes, int n_in,
                              void* d_out, int out_size, void* d_ws, size_t ws_size,
                              hipStream_t stream) {
  (void)in_sizes; (void)n_in; (void)out_size; (void)ws_size;
  const int Bq = 2048, Fc = 5;

  const int*   query    = (const int*)d_in[0];
  const int*   support  = (const int*)d_in[1];
  const int*   q_l_conn = (const int*)d_in[2];
  const int*   q_r_conn = (const int*)d_in[4];
  const int*   s_l_conn = (const int*)d_in[6];
  const int*   s_r_conn = (const int*)d_in[8];
  const float* emb      = (const float*)d_in[10];
  const float* gcn_w    = (const float*)d_in[11];
  const float* gcn_wb   = (const float*)d_in[12];
  const float* gcn_b    = (const float*)d_in[13];
  const float* p1_w     = (const float*)d_in[14];
  const float* p1_b     = (const float*)d_in[15];
  const float* p2_w     = (const float*)d_in[16];
  const float* p2_b     = (const float*)d_in[17];
  const float* ln_g     = (const float*)d_in[18];
  const float* ln_b     = (const float*)d_in[19];
  const float* w_ih     = (const float*)d_in[20];
  const float* w_hh     = (const float*)d_in[21];
  const float* b_ih     = (const float*)d_in[22];
  const float* b_hh     = (const float*)d_in[23];
  float* out = (float*)d_out;

  char* ws = (char*)d_ws;
  size_t off = 0;
  auto alloc = [&](size_t bytes) -> void* {
    void* p = ws + off;
    off = (off + bytes + 255) & ~(size_t)255;
    return p;
  };

  _Float16* gcn_w16 = (_Float16*)alloc((size_t)256 * 512 * 2);
  _Float16* p1_w16  = (_Float16*)alloc((size_t)1024 * 512 * 2);
  _Float16* p2_w16  = (_Float16*)alloc((size_t)512 * 1024 * 2);
  _Float16* w_ih16  = (_Float16*)alloc((size_t)4096 * 512 * 2);
  _Float16* w_hh16  = (_Float16*)alloc((size_t)4096 * 1024 * 2);
  float* gcn_bias   = (float*)alloc(256 * 4);
  float* lstm_bias  = (float*)alloc(4096 * 4);
  float* cat_q0     = (float*)alloc((size_t)Bq * 512 * 4);
  float* cat_q1     = (float*)alloc((size_t)Bq * 512 * 4);
  float* cat_s0     = (float*)alloc((size_t)Fc * 512 * 4);
  float* cat_s1     = (float*)alloc((size_t)Fc * 512 * 4);
  float* query_n    = (float*)alloc((size_t)Bq * 512 * 4);
  float* supp_n     = (float*)alloc((size_t)Fc * 512 * 4);
  float* enc_h      = (float*)alloc((size_t)Bq * 1024 * 4);  // reused q & s
  float* enc_o      = (float*)alloc((size_t)Bq * 512 * 4);   // reused q & s
  float* sup_enc    = (float*)alloc((size_t)Fc * 512 * 4);
  float* query_g    = (float*)alloc((size_t)Bq * 512 * 4);
  float* support_g  = (float*)alloc(512 * 4);
  float* gates_base = (float*)alloc((size_t)Bq * 4096 * 4);
  float* gates      = (float*)alloc((size_t)Bq * 4096 * 4);
  float* cstate     = (float*)alloc((size_t)Bq * 1024 * 4);
  float* hbuf       = (float*)alloc((size_t)Bq * 512 * 4);
  float* h_r        = (float*)alloc((size_t)Bq * 1024 * 4);

  auto cvt = [&](const float* src, _Float16* dst, int n) {
    cvt_f32_to_f16_kernel<<<(n + 255) / 256, 256, 0, stream>>>(src, dst, n);
  };
  cvt(gcn_w, gcn_w16, 256 * 512);
  cvt(p1_w,  p1_w16,  1024 * 512);
  cvt(p2_w,  p2_w16,  512 * 1024);
  cvt(w_ih,  w_ih16,  4096 * 512);
  cvt(w_hh,  w_hh16,  4096 * 1024);
  vec_add_kernel<<<1, 256, 0, stream>>>(gcn_wb, gcn_b, gcn_bias, 256);
  vec_add_kernel<<<16, 256, 0, stream>>>(b_ih, b_hh, lstm_bias, 4096);

  // neighbor encoders (gather + top-32 + mean): HBM-bound
  neighbor_mean_kernel<<<Bq, 256, 0, stream>>>(q_l_conn, query,   2, 0, emb, cat_q0);
  neighbor_mean_kernel<<<Bq, 256, 0, stream>>>(q_r_conn, query,   2, 1, emb, cat_q1);
  neighbor_mean_kernel<<<Fc, 256, 0, stream>>>(s_l_conn, support, 2, 0, emb, cat_s0);
  neighbor_mean_kernel<<<Fc, 256, 0, stream>>>(s_r_conn, support, 2, 1, emb, cat_s1);

  // GCN on the K-mean (linearity), fused bias+tanh, writing [ql|qr] halves
  launch_gemm(cat_q0, 512, gcn_w16, 512, query_n,       512, gcn_bias, nullptr, 0, Bq, 256, 512, 2, stream);
  launch_gemm(cat_q1, 512, gcn_w16, 512, query_n + 256, 512, gcn_bias, nullptr, 0, Bq, 256, 512, 2, stream);
  launch_gemm(cat_s0, 512, gcn_w16, 512, supp_n,        512, gcn_bias, nullptr, 0, Fc, 256, 512, 2, stream);
  launch_gemm(cat_s1, 512, gcn_w16, 512, supp_n + 256,  512, gcn_bias, nullptr, 0, Fc, 256, 512, 2, stream);

  // support encoder: supports -> support_g
  launch_gemm(supp_n, 512, p1_w16, 512, enc_h, 1024, p1_b, nullptr, 0, Fc, 1024, 512, 1, stream);
  launch_gemm(enc_h, 1024, p2_w16, 1024, enc_o, 512, p2_b, supp_n, 512, Fc, 512, 1024, 0, stream);
  layernorm_kernel<<<Fc, 256, 0, stream>>>(enc_o, sup_enc, ln_g, ln_b, 512);
  mean_rows_kernel<<<2, 256, 0, stream>>>(sup_enc, support_g, Fc, 512);

  // support encoder: queries -> query_g
  launch_gemm(query_n, 512, p1_w16, 512, enc_h, 1024, p1_b, nullptr, 0, Bq, 1024, 512, 1, stream);
  launch_gemm(enc_h, 1024, p2_w16, 1024, enc_o, 512, p2_b, query_n, 512, Bq, 512, 1024, 0, stream);
  layernorm_kernel<<<Bq, 256, 0, stream>>>(enc_o, query_g, ln_g, ln_b, 512);

  // LSTM match loop: hoist query_g @ w_ih.T (+ b_ih + b_hh); step 0 has
  // h_r == 0 so gates == gates_base (GEMM elided).
  launch_gemm(query_g, 512, w_ih16, 512, gates_base, 4096, lstm_bias, nullptr, 0, Bq, 4096, 512, 0, stream);
  zero_kernel<<<(Bq * 1024 + 255) / 256, 256, 0, stream>>>(cstate, Bq * 1024);
  lstm_cell_kernel<<<(Bq * 1024 + 255) / 256, 256, 0, stream>>>(
      gates_base, query_g, support_g, cstate, hbuf, h_r, Bq);
  for (int s = 1; s < 4; s++) {
    launch_gemm(h_r, 1024, w_hh16, 1024, gates, 4096, nullptr, gates_base, 4096,
                Bq, 4096, 1024, 0, stream);
    lstm_cell_kernel<<<(Bq * 1024 + 255) / 256, 256, 0, stream>>>(
        gates, query_g, support_g, cstate, hbuf, h_r, Bq);
  }

  // cosine(h, support_g)
  final_dot_kernel<<<Bq, 256, 0, stream>>>(hbuf, support_g, out);
}
